// GraphModel_68599217651878
// MI455X (gfx1250) — matrix operand order repaired and checked by reading, not compiled, not run on hardware
//
#include <hip/hip_runtime.h>
#include <hip/hip_bf16.h>

// GraphNet block for MI455X (gfx1250, wave32).
// edge MLP (384->128->128,LN) -> receiver segment-sum -> node MLP (320->128->128,LN)
// -> per-graph aggregates -> global MLP (320->128->128,LN).
// All GEMMs via v_wmma_f32_16x16x32_bf16 (bf16 in, f32 accumulate).

#define N_GRAPHS  64
#define NODES_PER 1024
#define EDGES_PER 8192
#define NTOT      (N_GRAPHS * NODES_PER)   // 65536
#define ETOT      (N_GRAPHS * EDGES_PER)   // 524288

typedef __attribute__((ext_vector_type(16))) __bf16 v16bf;
typedef __attribute__((ext_vector_type(8)))  float  v8f;

union Frag { v16bf v; uint4 q[2]; };

__device__ __forceinline__ unsigned short f2bf(float f) {
  // round-to-nearest-even f32 -> bf16
  unsigned u = __float_as_uint(f);
  return (unsigned short)((u + 0x7FFFu + ((u >> 16) & 1u)) >> 16);
}

__device__ __forceinline__ uint2 pack4bf(float4 v) {
  uint2 r;
  r.x = (unsigned)f2bf(v.x) | ((unsigned)f2bf(v.y) << 16);
  r.y = (unsigned)f2bf(v.z) | ((unsigned)f2bf(v.w) << 16);
  return r;
}

// One wave computes a 16x128 tile: rows [rowBase, rowBase+16), 128 output cols
// as 8 WMMA n-tiles. A is bf16 in LDS (row-major, stride lda halves, 16B
// aligned); W is bf16 pre-packed in B-fragment order; bias folded into acc.
// All 8 B fragments of a K-chunk are loaded into distinct registers first so
// the 8 WMMAs issue back-to-back behind a single wait; KT is compile-time so
// partial unrolling lets loads of chunk kc+1 overlap WMMAs of chunk kc.
template<int KT>
__device__ __forceinline__ void gemm_tile(
    const unsigned short* lA, int lda, int rowBase,
    const unsigned short* wpk, const float* bias,
    v8f acc[8], int lane)
{
  const int m = lane & 15;   // row within wave tile / column within n-tile
  const int h = lane >> 4;   // lane-half selects K sub-block
#pragma unroll
  for (int nt = 0; nt < 8; ++nt) {
    float b = bias[nt * 16 + m];
    v8f a;
#pragma unroll
    for (int i = 0; i < 8; ++i) a[i] = b;
    acc[nt] = a;
  }
#pragma unroll 2
  for (int kc = 0; kc < KT; ++kc) {
    // A fragment: 16-bit A 16x32 layout -> halves [8h,8h+8) and [16+8h,16+8h+8)
    Frag af;
    const unsigned short* ar = lA + (rowBase + m) * lda + kc * 32;
    af.q[0] = *(const uint4*)(ar + 8 * h);
    af.q[1] = *(const uint4*)(ar + 16 + 8 * h);
    // all 8 B fragments up front (one clause, distinct regs)
    Frag bf[8];
    const unsigned short* bp = wpk + ((size_t)kc * 8 * 32 + lane) * 16;
#pragma unroll
    for (int nt = 0; nt < 8; ++nt) {
      bf[nt].q[0] = *(const uint4*)(bp + (size_t)nt * 32 * 16);
      bf[nt].q[1] = *(const uint4*)(bp + (size_t)nt * 32 * 16 + 8);
    }
    if (kc + 1 < KT)
      __builtin_prefetch(wpk + (size_t)(kc + 1) * 8 * 32 * 16, 0, 3);
#pragma unroll
    for (int nt = 0; nt < 8; ++nt)
      acc[nt] = __builtin_amdgcn_wmma_f32_16x16x32_bf16(
          false, af.v, false, bf[nt].v, (short)0, acc[nt], false, false);
  }
}

// C/D layout: VGPR r, lanes 0-15 -> (M=r, N=lane), lanes 16-31 -> (M=r+8, N=lane-16)
__device__ __forceinline__ void store_relu_bf16(
    unsigned short* lH, int ldh, int rowBase, const v8f acc[8], int lane)
{
  const int m = lane & 15, h = lane >> 4;
#pragma unroll
  for (int nt = 0; nt < 8; ++nt)
#pragma unroll
    for (int r = 0; r < 8; ++r) {
      float v = acc[nt][r];
      v = v > 0.f ? v : 0.f;
      lH[(rowBase + r + 8 * h) * ldh + nt * 16 + m] = f2bf(v);
    }
}

__device__ __forceinline__ void store_f32(
    float* lF, int ldf, int rowBase, const v8f acc[8], int lane)
{
  const int m = lane & 15, h = lane >> 4;
#pragma unroll
  for (int nt = 0; nt < 8; ++nt)
#pragma unroll
    for (int r = 0; r < 8; ++r)
      lF[(rowBase + r + 8 * h) * ldf + nt * 16 + m] = acc[nt][r];
}

// Convert W[K x 128] f32 row-major into WMMA B-fragment order (bf16):
// fragment f = kc*8+nt; element j of lane -> W[kc*32 + 8*(lane/16) + j + (j>=8 ? 8 : 0)][nt*16 + lane%16]
__global__ void __launch_bounds__(256)
pack_weights_kernel(const float* __restrict__ W,
                    unsigned short* __restrict__ out, int K)
{
  int tid = blockIdx.x * blockDim.x + threadIdx.x;
  int nf = (K / 32) * 8;
  int f = tid >> 5;
  if (f >= nf) return;
  int lane = tid & 31;
  int kc = f >> 3, nt = f & 7;
  int col = nt * 16 + (lane & 15);
  int h = lane >> 4;
  Frag o;
  unsigned short* os = (unsigned short*)&o;
#pragma unroll
  for (int j = 0; j < 16; ++j) {
    int kk = kc * 32 + 8 * h + j + (j >= 8 ? 8 : 0);
    os[j] = f2bf(W[(size_t)kk * 128 + col]);
  }
  uint4* dst = (uint4*)(out + ((size_t)f * 32 + lane) * 16);
  dst[0] = o.q[0];
  dst[1] = o.q[1];
}

// ---------------- edge MLP: 128 edges per block, 256 threads (8 waves) ----------------
__global__ void __launch_bounds__(256)
edge_mlp_kernel(
    const float* __restrict__ nodes, const float* __restrict__ edges,
    const float* __restrict__ globals_, const int* __restrict__ receivers,
    const int* __restrict__ senders,
    const unsigned short* __restrict__ w1p, const float* __restrict__ b1,
    const unsigned short* __restrict__ w2p, const float* __restrict__ b2,
    const float* __restrict__ gamma, const float* __restrict__ beta,
    float* __restrict__ edges2, float* __restrict__ agg,
    float* __restrict__ edges_agg)
{
  extern __shared__ char smem[];
  unsigned short* lA = (unsigned short*)smem;              // 128 x 392 bf16 = 100352 B
  unsigned short* lH = (unsigned short*)(smem + 100352);   // 128 x 136 bf16 =  34816 B
  float* colsum      = (float*)(smem + 135168);            // 128 f32
  float* lF          = (float*)smem;                       // 128 x 132 f32 (aliases lA)

  const int tid = threadIdx.x;
  const int E0  = blockIdx.x * 128;
  const int gid = E0 / EDGES_PER;   // all 128 edges in one graph (8192 % 128 == 0)

  if (tid < 128) colsum[tid] = 0.f;

  // gather + f32->bf16 in float4 units: concat boundaries (128,256,320,384)
  // are all multiples of 4 -> c4 boundaries 32,64,80,96.
  for (int idx = tid; idx < 128 * 96; idx += 256) {
    int row = idx / 96, c4 = idx - row * 96;
    int e = E0 + row;
    float4 v;
    if (c4 < 32)      v = ((const float4*)(nodes + (size_t)receivers[e] * 128))[c4];
    else if (c4 < 64) v = ((const float4*)(nodes + (size_t)senders[e] * 128))[c4 - 32];
    else if (c4 < 80) v = ((const float4*)(edges + (size_t)e * 64))[c4 - 64];
    else              v = ((const float4*)(globals_ + gid * 64))[c4 - 80];
    *(uint2*)(lA + row * 392 + c4 * 4) = pack4bf(v);
  }
  __syncthreads();

  const int wave = tid >> 5, lane = tid & 31;
  const int rowBase = wave * 16;
  v8f acc[8];
  gemm_tile<12>(lA, 392, rowBase, w1p, b1, acc, lane);   // 384 = 12 K-chunks
  store_relu_bf16(lH, 136, rowBase, acc, lane);
  __syncthreads();
  gemm_tile<4>(lH, 136, rowBase, w2p, b2, acc, lane);    // 128 = 4 K-chunks
  store_f32(lF, 132, rowBase, acc, lane);                // lA dead after barrier above
  __syncthreads();

  if (tid < 128) {  // one row per thread: LayerNorm + scatter + column sums
    int row = tid;
    const float4* rp = (const float4*)(lF + row * 132);
    float s = 0.f, s2 = 0.f;
    for (int c4 = 0; c4 < 32; ++c4) {
      float4 x = rp[c4];
      s  += x.x + x.y + x.z + x.w;
      s2 += x.x * x.x + x.y * x.y + x.z * x.z + x.w * x.w;
    }
    float mean = s * (1.f / 128.f);
    float var  = s2 * (1.f / 128.f) - mean * mean;
    float rstd = rsqrtf(var + 1e-5f);
    int e  = E0 + row;
    int rc = receivers[e];
    float4* outp = (float4*)(edges2 + (size_t)e * 128);
    float* aggp  = agg + (size_t)rc * 128;
    const float4* gp = (const float4*)gamma;
    const float4* bp = (const float4*)beta;
    for (int c4 = 0; c4 < 32; ++c4) {
      float4 x = rp[c4], g = gp[c4], b = bp[c4], y;
      y.x = (x.x - mean) * rstd * g.x + b.x;
      y.y = (x.y - mean) * rstd * g.y + b.y;
      y.z = (x.z - mean) * rstd * g.z + b.z;
      y.w = (x.w - mean) * rstd * g.w + b.w;
      outp[c4] = y;
      atomicAdd(aggp + c4 * 4 + 0, y.x);   // segment_sum over receivers
      atomicAdd(aggp + c4 * 4 + 1, y.y);
      atomicAdd(aggp + c4 * 4 + 2, y.z);
      atomicAdd(aggp + c4 * 4 + 3, y.w);
      atomicAdd(colsum + c4 * 4 + 0, y.x); // per-graph edge aggregate (LDS)
      atomicAdd(colsum + c4 * 4 + 1, y.y);
      atomicAdd(colsum + c4 * 4 + 2, y.z);
      atomicAdd(colsum + c4 * 4 + 3, y.w);
    }
  }
  __syncthreads();
  if (tid < 128) atomicAdd(edges_agg + gid * 128 + tid, colsum[tid]);
}

// ---------------- node MLP: 128 nodes per block ----------------
__global__ void __launch_bounds__(256)
node_mlp_kernel(
    const float* __restrict__ nodes, const float* __restrict__ agg,
    const float* __restrict__ globals_,
    const unsigned short* __restrict__ w1p, const float* __restrict__ b1,
    const unsigned short* __restrict__ w2p, const float* __restrict__ b2,
    const float* __restrict__ gamma, const float* __restrict__ beta,
    float* __restrict__ nodes2, float* __restrict__ nodes_agg)
{
  extern __shared__ char smem[];
  unsigned short* lA = (unsigned short*)smem;              // 128 x 328 bf16 = 83968 B
  unsigned short* lH = (unsigned short*)(smem + 83968);    // 128 x 136 bf16 = 34816 B
  float* colsum      = (float*)(smem + 118784);            // 128 f32
  float* lF          = (float*)smem;                       // 128 x 132 f32 (aliases lA)

  const int tid = threadIdx.x;
  const int N0  = blockIdx.x * 128;
  const int gid = N0 / NODES_PER;   // 1024 % 128 == 0

  if (tid < 128) colsum[tid] = 0.f;

  // concat boundaries (128,256,320) -> c4 boundaries 32,64,80
  for (int idx = tid; idx < 128 * 80; idx += 256) {
    int row = idx / 80, c4 = idx - row * 80;
    int n = N0 + row;
    float4 v;
    if (c4 < 32)      v = ((const float4*)(nodes + (size_t)n * 128))[c4];
    else if (c4 < 64) v = ((const float4*)(agg + (size_t)n * 128))[c4 - 32];
    else              v = ((const float4*)(globals_ + gid * 64))[c4 - 64];
    *(uint2*)(lA + row * 328 + c4 * 4) = pack4bf(v);
  }
  __syncthreads();

  const int wave = tid >> 5, lane = tid & 31;
  const int rowBase = wave * 16;
  v8f acc[8];
  gemm_tile<10>(lA, 328, rowBase, w1p, b1, acc, lane);   // 320 = 10 K-chunks
  store_relu_bf16(lH, 136, rowBase, acc, lane);
  __syncthreads();
  gemm_tile<4>(lH, 136, rowBase, w2p, b2, acc, lane);
  store_f32(lF, 132, rowBase, acc, lane);
  __syncthreads();

  if (tid < 128) {
    int row = tid;
    const float4* rp = (const float4*)(lF + row * 132);
    float s = 0.f, s2 = 0.f;
    for (int c4 = 0; c4 < 32; ++c4) {
      float4 x = rp[c4];
      s  += x.x + x.y + x.z + x.w;
      s2 += x.x * x.x + x.y * x.y + x.z * x.z + x.w * x.w;
    }
    float mean = s * (1.f / 128.f);
    float var  = s2 * (1.f / 128.f) - mean * mean;
    float rstd = rsqrtf(var + 1e-5f);
    float4* outp = (float4*)(nodes2 + (size_t)(N0 + row) * 128);
    const float4* gp = (const float4*)gamma;
    const float4* bp = (const float4*)beta;
    for (int c4 = 0; c4 < 32; ++c4) {
      float4 x = rp[c4], g = gp[c4], b = bp[c4], y;
      y.x = (x.x - mean) * rstd * g.x + b.x;
      y.y = (x.y - mean) * rstd * g.y + b.y;
      y.z = (x.z - mean) * rstd * g.z + b.z;
      y.w = (x.w - mean) * rstd * g.w + b.w;
      outp[c4] = y;
      atomicAdd(colsum + c4 * 4 + 0, y.x);
      atomicAdd(colsum + c4 * 4 + 1, y.y);
      atomicAdd(colsum + c4 * 4 + 2, y.z);
      atomicAdd(colsum + c4 * 4 + 3, y.w);
    }
  }
  __syncthreads();
  if (tid < 128) atomicAdd(nodes_agg + gid * 128 + tid, colsum[tid]);
}

// ---------------- global MLP: 64 graphs, one block of 128 threads (4 waves) ----------------
__global__ void __launch_bounds__(128)
global_mlp_kernel(
    const float* __restrict__ globals_, const float* __restrict__ nodes_agg,
    const float* __restrict__ edges_agg,
    const unsigned short* __restrict__ w1p, const float* __restrict__ b1,
    const unsigned short* __restrict__ w2p, const float* __restrict__ b2,
    const float* __restrict__ gamma, const float* __restrict__ beta,
    float* __restrict__ out)
{
  extern __shared__ char smem[];
  unsigned short* lA = (unsigned short*)smem;              // 64 x 328 bf16 = 41984 B
  unsigned short* lH = (unsigned short*)(smem + 41984);    // 64 x 136 bf16 = 17408 B
  float* lF          = (float*)smem;                       // 64 x 132 f32 (aliases lA)

  const int tid = threadIdx.x;
  // concat boundaries (64,192,320) -> c4 boundaries 16,48,80
  for (int idx = tid; idx < 64 * 80; idx += 128) {
    int row = idx / 80, c4 = idx - row * 80;
    float4 v;
    if (c4 < 16)      v = ((const float4*)(globals_ + row * 64))[c4];
    else if (c4 < 48) v = ((const float4*)(nodes_agg + row * 128))[c4 - 16];
    else              v = ((const float4*)(edges_agg + row * 128))[c4 - 48];
    *(uint2*)(lA + row * 328 + c4 * 4) = pack4bf(v);
  }
  __syncthreads();

  const int wave = tid >> 5, lane = tid & 31;
  const int rowBase = wave * 16;   // 4 waves cover M=64
  v8f acc[8];
  gemm_tile<10>(lA, 328, rowBase, w1p, b1, acc, lane);
  store_relu_bf16(lH, 136, rowBase, acc, lane);
  __syncthreads();
  gemm_tile<4>(lH, 136, rowBase, w2p, b2, acc, lane);
  store_f32(lF, 132, rowBase, acc, lane);
  __syncthreads();

  if (tid < 64) {
    int row = tid;
    const float4* rp = (const float4*)(lF + row * 132);
    float s = 0.f, s2 = 0.f;
    for (int c4 = 0; c4 < 32; ++c4) {
      float4 x = rp[c4];
      s  += x.x + x.y + x.z + x.w;
      s2 += x.x * x.x + x.y * x.y + x.z * x.z + x.w * x.w;
    }
    float mean = s * (1.f / 128.f);
    float var  = s2 * (1.f / 128.f) - mean * mean;
    float rstd = rsqrtf(var + 1e-5f);
    float4* outp = (float4*)(out + row * 128);
    const float4* gp = (const float4*)gamma;
    const float4* bp = (const float4*)beta;
    for (int c4 = 0; c4 < 32; ++c4) {
      float4 x = rp[c4], g = gp[c4], b = bp[c4], y;
      y.x = (x.x - mean) * rstd * g.x + b.x;
      y.y = (x.y - mean) * rstd * g.y + b.y;
      y.z = (x.z - mean) * rstd * g.z + b.z;
      y.w = (x.w - mean) * rstd * g.w + b.w;
      outp[c4] = y;
    }
  }
}

extern "C" void kernel_launch(void* const* d_in, const int* in_sizes, int n_in,
                              void* d_out, int out_size, void* d_ws, size_t ws_size,
                              hipStream_t stream)
{
  const float* nodes    = (const float*)d_in[0];
  const float* edges    = (const float*)d_in[1];
  const float* globals_ = (const float*)d_in[2];
  const int* receivers  = (const int*)d_in[3];
  const int* senders    = (const int*)d_in[4];
  const float* e_w1 = (const float*)d_in[5];  const float* e_b1 = (const float*)d_in[6];
  const float* e_w2 = (const float*)d_in[7];  const float* e_b2 = (const float*)d_in[8];
  const float* e_g  = (const float*)d_in[9];  const float* e_bt = (const float*)d_in[10];
  const float* n_w1 = (const float*)d_in[11]; const float* n_b1 = (const float*)d_in[12];
  const float* n_w2 = (const float*)d_in[13]; const float* n_b2 = (const float*)d_in[14];
  const float* n_g  = (const float*)d_in[15]; const float* n_bt = (const float*)d_in[16];
  const float* g_w1 = (const float*)d_in[17]; const float* g_b1 = (const float*)d_in[18];
  const float* g_w2 = (const float*)d_in[19]; const float* g_b2 = (const float*)d_in[20];
  const float* g_g  = (const float*)d_in[21]; const float* g_bt = (const float*)d_in[22];

  char* ws = (char*)d_ws;
  float* agg        = (float*)(ws + 0);           // N*128 f32 = 33554432 B
  float* nodes_agg  = (float*)(ws + 33554432);    // 64*128 f32
  float* edges_agg  = (float*)(ws + 33587200);    // 64*128 f32
  unsigned short* e_w1p = (unsigned short*)(ws + 33619968); // 96 frags * 512 halves
  unsigned short* e_w2p = (unsigned short*)(ws + 33718272); // 32 frags
  unsigned short* n_w1p = (unsigned short*)(ws + 33751040); // 80 frags
  unsigned short* n_w2p = (unsigned short*)(ws + 33832960); // 32 frags
  unsigned short* g_w1p = (unsigned short*)(ws + 33865728); // 80 frags
  unsigned short* g_w2p = (unsigned short*)(ws + 33947648); // 32 frags

  // accumulators must be zero every launch (agg / nodes_agg / edges_agg contiguous)
  hipMemsetAsync(agg, 0, 33554432 + 32768 + 32768, stream);

  pack_weights_kernel<<<12, 256, 0, stream>>>(e_w1, e_w1p, 384);
  pack_weights_kernel<<<4,  256, 0, stream>>>(e_w2, e_w2p, 128);
  pack_weights_kernel<<<10, 256, 0, stream>>>(n_w1, n_w1p, 320);
  pack_weights_kernel<<<4,  256, 0, stream>>>(n_w2, n_w2p, 128);
  pack_weights_kernel<<<10, 256, 0, stream>>>(g_w1, g_w1p, 320);
  pack_weights_kernel<<<4,  256, 0, stream>>>(g_w2, g_w2p, 128);

  float* out_nodes2 = (float*)d_out;                        // [N,128]
  float* out_edges2 = out_nodes2 + (size_t)NTOT * 128;      // [E,128]
  float* out_glob2  = out_edges2 + (size_t)ETOT * 128;      // [64,128]

  edge_mlp_kernel<<<ETOT / 128, 256, 135680, stream>>>(
      nodes, edges, globals_, receivers, senders,
      e_w1p, e_b1, e_w2p, e_b2, e_g, e_bt,
      out_edges2, agg, edges_agg);

  node_mlp_kernel<<<NTOT / 128, 256, 119296, stream>>>(
      nodes, agg, globals_,
      n_w1p, n_b1, n_w2p, n_b2, n_g, n_bt,
      out_nodes2, nodes_agg);

  global_mlp_kernel<<<1, 128, 59392, stream>>>(
      globals_, nodes_agg, edges_agg,
      g_w1p, g_b1, g_w2p, g_b2, g_g, g_bt,
      out_glob2);
}
